// MelSpectrogramNet_30751965839793
// MI455X (gfx1250) — compile-verified, matrix operand order checked by reading, not run on hardware
//
#include <hip/hip_runtime.h>
#include <hip/hip_bf16.h>
#include <math.h>

typedef __attribute__((ext_vector_type(16))) _Float16 v16h;
typedef __attribute__((ext_vector_type(8)))  _Float16 v8h;
typedef __attribute__((ext_vector_type(8)))  float    v8f;

// ---------------- problem constants ----------------
static constexpr int    NB     = 32;        // batch
static constexpr int    TLEN   = 221184;    // samples per batch
static constexpr int    HOP    = 256;
static constexpr int    WIN    = 2048;
static constexpr int    NFRAME = 857;       // (221184-2048)/256+1
static constexpr int    MROWS  = NB * NFRAME;      // 27424 = 1714*16
static constexpr int    MT     = MROWS / 16;       // 1714 M tiles
static constexpr int    NMQ    = 4;                // M-tiles per wave (stage 1)
static constexpr int    MG     = (MT + NMQ - 1) / NMQ;  // 429 padded M groups
static constexpr int    NFREQ  = 1025;
static constexpr int    FT     = 66;               // freq tiles (pad to 1056)
static constexpr int    FPAD   = FT * 16;          // 1056
static constexpr int    KC1    = WIN / 32;         // 64 K-chunks, stage 1
static constexpr int    KC2    = FPAD / 32;        // 33 K-chunks, stage 2
static constexpr int    NMEL   = 128;
static constexpr int    NT2    = NMEL / 16;        // 8 mel N-tiles

// ---------------- workspace layout (bytes) ----------------
static constexpr size_t OFF_XH   = 0;                                   // f16 x
static constexpr size_t SZ_XH    = (size_t)NB * TLEN * 2;               // 14,155,776
static constexpr size_t OFF_W1   = OFF_XH + SZ_XH;
static constexpr size_t SZ_W1    = (size_t)2 * FT * KC1 * 512 * 2;      //  8,650,752
static constexpr size_t OFF_WM   = OFF_W1 + SZ_W1;
static constexpr size_t SZ_WM    = (size_t)NT2 * KC2 * 512 * 2;         //    270,336
static constexpr size_t OFF_P    = OFF_WM + SZ_WM;
static constexpr size_t SZ_P     = (size_t)MROWS * FPAD * 2;            // 57,919,488
static constexpr size_t OFF_MEL  = OFF_P + SZ_P;
static constexpr size_t SZ_MEL   = (size_t)MROWS * NMEL * 4;            // 14,041,088
static constexpr size_t OFF_GMAX = OFF_MEL + SZ_MEL;                    // ~95.0 MB total

// K index inside a 32-wide chunk for element j (0..15) of lane l, per the
// CDNA5 16-bit A/B VGPR layout (lanes 0-15: K 0-7,16-23; lanes 16-31: K 8-15,24-31)
__device__ __forceinline__ int kmap(int lane, int j) {
    return ((lane >> 4) * 8) + (j & 7) + ((j >= 8) ? 16 : 0);
}

__global__ void k_init(unsigned int* gmax) { *gmax = 0u; }

__global__ void k_tohalf(const float* __restrict__ x, _Float16* __restrict__ xh, int n) {
    int i = blockIdx.x * blockDim.x + threadIdx.x;
    if (i < n) xh[i] = (_Float16)x[i];
}

// Pack cos/sin weights [1025][2048] f32 into B-tile blobs:
// blob ((p*FT+nt)*KC1+kc): 32 lanes x 16 halves, element = W[nt*16+(l&15)][kc*32+kmap]
__global__ void k_packW1(const float* __restrict__ cw, const float* __restrict__ sw,
                         _Float16* __restrict__ dst, int n) {
    int i = blockIdx.x * blockDim.x + threadIdx.x;
    if (i >= n) return;
    int j    = i & 15;
    int lane = (i >> 4) & 31;
    int kc   = (i >> 9) & 63;
    int rest = i >> 15;
    int nt   = rest % FT;
    int p    = rest / FT;
    int f = nt * 16 + (lane & 15);
    int k = kc * 32 + kmap(lane, j);
    const float* src = p ? sw : cw;
    float v = (f < NFREQ) ? src[(size_t)f * WIN + k] : 0.0f;
    dst[i] = (_Float16)v;
}

// Pack mel weights [128][1025] into B-tile blobs over K=freq (padded to 1056)
__global__ void k_packWm(const float* __restrict__ mw, _Float16* __restrict__ dst, int n) {
    int i = blockIdx.x * blockDim.x + threadIdx.x;
    if (i >= n) return;
    int j    = i & 15;
    int lane = (i >> 4) & 31;
    int kc   = (i >> 9) % KC2;
    int nt   = (i >> 9) / KC2;
    int m = nt * 16 + (lane & 15);
    int k = kc * 32 + kmap(lane, j);
    float v = (k < NFREQ) ? mw[(size_t)m * NFREQ + k] : 0.0f;
    dst[i] = (_Float16)v;
}

// Stage 1: conv-DFT + power. One wave per (4 M-tiles x 2 freq-tiles).
// 16 accumulators (4 x 2 x cos/sin), 16 v_wmma per K-chunk, 64 chunks.
__global__ __launch_bounds__(32)
void k_gemm1(const _Float16* __restrict__ xh, const _Float16* __restrict__ W1,
             _Float16* __restrict__ P) {
    const int l  = threadIdx.x;
    const int lm = l & 15;
    const int lk = (l >> 4) * 8;
    const int mg = blockIdx.x;   // 0..428 (4 M-tiles each; last group padded)
    const int fp = blockIdx.y;   // 0..32  (2 freq tiles each)

    long aoff[NMQ];
    #pragma unroll
    for (int mi = 0; mi < NMQ; ++mi) {
        int m = (mg * NMQ + mi) * 16 + lm;     // global frame row for this lane
        if (m >= MROWS) m = MROWS - 1;         // clamp pad rows to a valid frame
        int b = m / NFRAME;
        int t = m - b * NFRAME;
        aoff[mi] = (long)b * TLEN + (long)t * HOP + lk;
    }

    v8f acc[NMQ][2][2] = {};  // [mi][fi][plane]
    for (int kc = 0; kc < KC1; ++kc) {
        v16h A[NMQ];
        #pragma unroll
        for (int mi = 0; mi < NMQ; ++mi) {
            const _Float16* pa = xh + aoff[mi] + kc * 32;
            __builtin_prefetch(pa + 64, 0, 3);              // kc+2
            v8h a0 = *(const v8h*)(pa);
            v8h a1 = *(const v8h*)(pa + 16);
            A[mi] = __builtin_shufflevector(a0, a1, 0,1,2,3,4,5,6,7,8,9,10,11,12,13,14,15);
        }
        #pragma unroll
        for (int fi = 0; fi < 2; ++fi) {
            #pragma unroll
            for (int p = 0; p < 2; ++p) {
                const _Float16* pb =
                    W1 + ((((size_t)p * FT + (fp * 2 + fi)) * KC1 + kc) * 32 + l) * 16;
                __builtin_prefetch(pb + 1024, 0, 3);        // kc+2 (chunks contiguous)
                v16h B = *(const v16h*)pb;
                #pragma unroll
                for (int mi = 0; mi < NMQ; ++mi)
                    acc[mi][fi][p] = __builtin_amdgcn_wmma_f32_16x16x32_f16(
                        false, A[mi], false, B, (short)0, acc[mi][fi][p], false, false);
            }
        }
    }

    // power = cos^2 + sin^2, store f16 into padded P[MROWS][FPAD]
    #pragma unroll
    for (int mi = 0; mi < NMQ; ++mi) {
        #pragma unroll
        for (int fi = 0; fi < 2; ++fi) {
            v8f pc = acc[mi][fi][0];
            v8f ps = acc[mi][fi][1];
            int colg = (fp * 2 + fi) * 16 + lm;
            int rowb = (mg * NMQ + mi) * 16 + (l >> 4) * 8;
            #pragma unroll
            for (int r = 0; r < 8; ++r) {
                float pw = pc[r] * pc[r] + ps[r] * ps[r];
                if (rowb + r < MROWS)
                    P[(size_t)(rowb + r) * FPAD + colg] = (_Float16)pw;
            }
        }
    }
}

// Stage 2: mel projection. One wave per M-tile computes all 128 mels.
__global__ __launch_bounds__(32)
void k_gemm2(const _Float16* __restrict__ P, const _Float16* __restrict__ Wm,
             float* __restrict__ mel, unsigned int* __restrict__ gmax) {
    const int l  = threadIdx.x;
    const int lm = l & 15;
    const int lk = (l >> 4) * 8;
    const int mt = blockIdx.x;  // 0..1713

    long aoff = (long)(mt * 16 + lm) * FPAD + lk;
    v8f acc[NT2] = {};
    for (int kc = 0; kc < KC2; ++kc) {
        const _Float16* pa = P + aoff + kc * 32;
        v8h a0 = *(const v8h*)(pa);
        v8h a1 = *(const v8h*)(pa + 16);
        v16h A = __builtin_shufflevector(a0, a1, 0,1,2,3,4,5,6,7,8,9,10,11,12,13,14,15);
        #pragma unroll
        for (int nt = 0; nt < NT2; ++nt) {
            const _Float16* pb = Wm + (((size_t)nt * KC2 + kc) * 32 + l) * 16;
            v16h B = *(const v16h*)pb;
            acc[nt] = __builtin_amdgcn_wmma_f32_16x16x32_f16(
                false, A, false, B, (short)0, acc[nt], false, false);
        }
    }

    float mx = 0.0f;
    int rowb = mt * 16 + (l >> 4) * 8;
    #pragma unroll
    for (int nt = 0; nt < NT2; ++nt) {
        #pragma unroll
        for (int r = 0; r < 8; ++r) {
            float v = acc[nt][r];
            mel[(size_t)(rowb + r) * NMEL + nt * 16 + lm] = v;
            mx = fmaxf(mx, v);
        }
    }
    // wave32 max-reduce, then global atomic (mel >= 0, so uint order == float order)
    #pragma unroll
    for (int off = 16; off >= 1; off >>= 1)
        mx = fmaxf(mx, __shfl_xor(mx, off, 32));
    if (l == 0) atomicMax(gmax, __float_as_uint(mx));
}

// Stage 3: dB + global top_db clamp + affine, reorder to [b][mel][t]
__global__ void k_final(const float* __restrict__ mel, const unsigned int* __restrict__ gmax,
                        float* __restrict__ out, int total) {
    int i = blockIdx.x * blockDim.x + threadIdx.x;
    if (i >= total) return;
    int t = i % NFRAME;
    int r = i / NFRAME;
    int m = r % NMEL;
    int b = r / NMEL;
    float v   = mel[((size_t)b * NFRAME + t) * NMEL + m];
    float gm  = __uint_as_float(*gmax);
    float ref = 10.0f * log10f(fmaxf(gm, 1e-10f));
    float db  = 10.0f * log10f(fmaxf(v, 1e-10f));
    db = fmaxf(db, ref - 80.0f);
    out[i] = (db + 25.0f) * (1.0f / 80.0f);
}

extern "C" void kernel_launch(void* const* d_in, const int* in_sizes, int n_in,
                              void* d_out, int out_size, void* d_ws, size_t ws_size,
                              hipStream_t stream) {
    const float* x    = (const float*)d_in[0];   // [32,1,221184]
    const float* cw   = (const float*)d_in[1];   // [1025,1,2048]
    const float* sw   = (const float*)d_in[2];   // [1025,1,2048]
    const float* mw   = (const float*)d_in[3];   // [128,1025]
    float*       out  = (float*)d_out;

    char* ws = (char*)d_ws;
    _Float16*     xh   = (_Float16*)(ws + OFF_XH);
    _Float16*     W1   = (_Float16*)(ws + OFF_W1);
    _Float16*     Wm   = (_Float16*)(ws + OFF_WM);
    _Float16*     P    = (_Float16*)(ws + OFF_P);
    float*        mel  = (float*)(ws + OFF_MEL);
    unsigned int* gmax = (unsigned int*)(ws + OFF_GMAX);

    k_init<<<1, 1, 0, stream>>>(gmax);

    {   int n = NB * TLEN;
        k_tohalf<<<(n + 255) / 256, 256, 0, stream>>>(x, xh, n); }

    {   int n = 2 * FT * KC1 * 512;
        k_packW1<<<(n + 255) / 256, 256, 0, stream>>>(cw, sw, W1, n); }

    {   int n = NT2 * KC2 * 512;
        k_packWm<<<(n + 255) / 256, 256, 0, stream>>>(mw, Wm, n); }

    {   dim3 grid(MG, FT / 2);   // 429 x 33
        k_gemm1<<<grid, 32, 0, stream>>>(xh, W1, P); }

    k_gemm2<<<MT, 32, 0, stream>>>(P, Wm, mel, gmax);

    {   int n = out_size;            // 32*128*857
        k_final<<<(n + 255) / 256, 256, 0, stream>>>(mel, gmax, out, n); }
    (void)in_sizes; (void)n_in; (void)ws_size;
}
